// SpatialTransformer_66511863546385
// MI455X (gfx1250) — matrix operand order repaired and checked by reading, not compiled, not run on hardware
//
#include <hip/hip_runtime.h>
#include <hip/hip_bf16.h>
#include <math.h>

// ---------------------------------------------------------------------------
// SpatialTransformer for MI455X (gfx1250), bf16 WMMA compute, f32 residual.
// GEMM: 32x32 tile per wave, 4x v_wmma_f32_16x16x32_bf16 per K-step,
// unconditional 128-bit fragment loads (index-clamped), store-side masking.
// ---------------------------------------------------------------------------

typedef __bf16 bf16_t;
typedef bf16_t v16bf __attribute__((ext_vector_type(16)));
typedef bf16_t v8bf  __attribute__((ext_vector_type(8)));
typedef float  v8f   __attribute__((ext_vector_type(8)));

#define BB   8
#define CC   512
#define HWD  1024
#define NHD  8
#define DHD  64
#define LL   2
#define NCD  77     // cond tokens
#define NCP  80     // cond rows padded to 16
#define NCS  96     // cross scores K padded to 32
#define DCD  768
#define F1D  4096
#define F2D  2048
#define MALL (BB*HWD)

// A fragment (16x32, row major): elems 0..15 -> K=+0..15
__device__ __forceinline__ v16bf ldfragA(const bf16_t* p)
{
  v8bf lo = *(const v8bf*)p;
  v8bf hi = *(const v8bf*)(p + 8);
  v16bf a;
#pragma unroll
  for (int j = 0; j < 8; ++j) { a[j] = lo[j]; a[j + 8] = hi[j]; }
  return a;
}
// B fragment from B^T (NxK row major): lane holds col, elems -> K=+0..15
__device__ __forceinline__ v16bf ldfragB(const bf16_t* p)
{
  v8bf lo = *(const v8bf*)p;
  v8bf hi = *(const v8bf*)(p + 8);
  v16bf b;
#pragma unroll
  for (int j = 0; j < 8; ++j) { b[j] = lo[j]; b[j + 8] = hi[j]; }
  return b;
}

#define WMMA_BF16(A_, B_, C_) \
  __builtin_amdgcn_wmma_f32_16x16x32_bf16(false, (A_), false, (B_), (short)0, (C_), false, false)

// D[z] = A[z] (MxK row-major bf16) * B^T[z] (NxK row-major bf16) + bias (+D if ACC)
// One wave computes a 32x32 tile (2x2 subtiles of 16x16). 8 waves / block.
template<bool OUT_F32, bool ACC>
__global__ __launch_bounds__(256) void gemm_wmma(
    const bf16_t* __restrict__ A, long lda, long strideA,
    const bf16_t* __restrict__ BT, long ldb, long strideB,
    void* __restrict__ Dv, long ldd, long strideD,
    const float* __restrict__ bias,
    int M, int N, int Kv)
{
  const int lane   = threadIdx.x & 31;
  const int wave   = threadIdx.x >> 5;
  const int tilesN = (N + 31) >> 5;
  const int tiles  = ((M + 31) >> 5) * tilesN;
  const int tile   = blockIdx.x * 8 + wave;
  if (tile >= tiles) return;                 // wave-uniform exit
  const int mt = tile / tilesN, nt = tile % tilesN;
  const int z  = blockIdx.z;

  A  += (long)z * strideA;
  BT += (long)z * strideB;
  float*  Df = (float*)Dv;
  bf16_t* Db = (bf16_t*)Dv;
  const long dbase = (long)z * strideD;

  const int l15  = lane & 15;
  const int half = lane >> 4;
  const int row0 = mt * 32 + l15, row1 = row0 + 16;
  const int col0 = nt * 32 + l15, col1 = col0 + 16;
  // clamp so loads are always in-bounds; clamped lanes only feed masked outputs
  const int rA0 = min(row0, M - 1), rA1 = min(row1, M - 1);
  const int cB0 = min(col0, N - 1), cB1 = min(col1, N - 1);
  const int kbA = half << 4;                 // A: K-base 0 / 16
  const int kbB = half << 4;                 // B^T: K-base 0 / 16
  const int kend = (Kv + 31) & ~31;

  const bf16_t* pA0 = A  + (long)rA0 * lda + kbA;
  const bf16_t* pA1 = A  + (long)rA1 * lda + kbA;
  const bf16_t* pB0 = BT + (long)cB0 * ldb + kbB;
  const bf16_t* pB1 = BT + (long)cB1 * ldb + kbB;

  v8f acc00 = {0.f,0.f,0.f,0.f,0.f,0.f,0.f,0.f};
  v8f acc01 = acc00, acc10 = acc00, acc11 = acc00;

  for (int k0 = 0; k0 < kend; k0 += 32) {
    const v16bf a0 = ldfragA(pA0 + k0);
    const v16bf a1 = ldfragA(pA1 + k0);
    const v16bf b0 = ldfragB(pB0 + k0);
    const v16bf b1 = ldfragB(pB1 + k0);
    acc00 = WMMA_BF16(a0, b0, acc00);
    acc01 = WMMA_BF16(a0, b1, acc01);
    acc10 = WMMA_BF16(a1, b0, acc10);
    acc11 = WMMA_BF16(a1, b1, acc11);
  }

  const float bb0 = (bias != nullptr) ? bias[cB0] : 0.0f;
  const float bb1 = (bias != nullptr) ? bias[cB1] : 0.0f;
  const bool c0ok = (col0 < N), c1ok = (col1 < N);
#pragma unroll
  for (int r = 0; r < 8; ++r) {
    const int rr0 = mt * 32 + r + (half << 3);   // D: vgpr r -> row r (+8 upper half)
    const int rr1 = rr0 + 16;
    if (rr0 < M) {
      if (c0ok) { const long i = dbase + (long)rr0 * ldd + col0; float v = acc00[r] + bb0;
                  if (ACC) v += Df[i]; if (OUT_F32) Df[i] = v; else Db[i] = (bf16_t)v; }
      if (c1ok) { const long i = dbase + (long)rr0 * ldd + col1; float v = acc01[r] + bb1;
                  if (ACC) v += Df[i]; if (OUT_F32) Df[i] = v; else Db[i] = (bf16_t)v; }
    }
    if (rr1 < M) {
      if (c0ok) { const long i = dbase + (long)rr1 * ldd + col0; float v = acc10[r] + bb0;
                  if (ACC) v += Df[i]; if (OUT_F32) Df[i] = v; else Db[i] = (bf16_t)v; }
      if (c1ok) { const long i = dbase + (long)rr1 * ldd + col1; float v = acc11[r] + bb1;
                  if (ACC) v += Df[i]; if (OUT_F32) Df[i] = v; else Db[i] = (bf16_t)v; }
    }
  }
}

// ----------------------------- normalization -------------------------------
__global__ __launch_bounds__(256) void layernorm_kernel(
    const float* __restrict__ x, const float* __restrict__ s,
    const float* __restrict__ b, bf16_t* __restrict__ y, int nc)
{
  __shared__ float red[256];
  const long row = blockIdx.x;
  const float* xr = x + row * nc;
  float sum = 0.f, sq = 0.f;
  for (int c = threadIdx.x; c < nc; c += 256) { float v = xr[c]; sum += v; sq += v * v; }
  red[threadIdx.x] = sum; __syncthreads();
  for (int o = 128; o; o >>= 1) { if (threadIdx.x < o) red[threadIdx.x] += red[threadIdx.x + o]; __syncthreads(); }
  const float mean = red[0] / nc; __syncthreads();
  red[threadIdx.x] = sq; __syncthreads();
  for (int o = 128; o; o >>= 1) { if (threadIdx.x < o) red[threadIdx.x] += red[threadIdx.x + o]; __syncthreads(); }
  const float var  = red[0] / nc - mean * mean;
  const float rstd = rsqrtf(var + 1e-5f);
  bf16_t* yr = y + row * nc;
  for (int c = threadIdx.x; c < nc; c += 256)
    yr[c] = (bf16_t)((xr[c] - mean) * rstd * s[c] + b[c]);
}

// GroupNorm over (B,C,H,W), 32 groups; writes normalized bf16 in token layout (B,HW,C)
__global__ __launch_bounds__(256) void groupnorm_kernel(
    const float* __restrict__ x, const float* __restrict__ gs,
    const float* __restrict__ gb, bf16_t* __restrict__ y)
{
  __shared__ float r1[256], r2[256];
  const int b = blockIdx.x >> 5, g = blockIdx.x & 31;
  const int CPG = CC / 32;
  float sum = 0.f, sq = 0.f;
  for (int i = threadIdx.x; i < CPG * HWD; i += 256) {
    const int c = g * CPG + (i >> 10), p = i & (HWD - 1);
    const float v = x[((long)(b * CC + c)) * HWD + p];
    sum += v; sq += v * v;
  }
  r1[threadIdx.x] = sum; r2[threadIdx.x] = sq; __syncthreads();
  for (int o = 128; o; o >>= 1) {
    if (threadIdx.x < o) { r1[threadIdx.x] += r1[threadIdx.x + o]; r2[threadIdx.x] += r2[threadIdx.x + o]; }
    __syncthreads();
  }
  const float n = (float)(CPG * HWD);
  const float mean = r1[0] / n, var = r2[0] / n - mean * mean;
  const float rstd = rsqrtf(var + 1e-6f);
  for (int i = threadIdx.x; i < CPG * HWD; i += 256) {
    const int c = g * CPG + (i >> 10), p = i & (HWD - 1);
    const float v = x[((long)(b * CC + c)) * HWD + p];
    y[((long)b * HWD + p) * CC + c] = (bf16_t)((v - mean) * rstd * gs[c] + gb[c]);
  }
}

// ------------------------------- softmax -----------------------------------
__global__ __launch_bounds__(256) void softmax_kernel(
    const float* __restrict__ S, bf16_t* __restrict__ P, int len, int pad, int ldS)
{
  __shared__ float red[256];
  const long row = blockIdx.x;
  const float* sr = S + row * ldS;
  bf16_t* pr = P + row * pad;
  float mx = -1e30f;
  for (int j = threadIdx.x; j < len; j += 256) mx = fmaxf(mx, sr[j]);
  red[threadIdx.x] = mx; __syncthreads();
  for (int o = 128; o; o >>= 1) { if (threadIdx.x < o) red[threadIdx.x] = fmaxf(red[threadIdx.x], red[threadIdx.x + o]); __syncthreads(); }
  mx = red[0]; __syncthreads();
  float sum = 0.f;
  for (int j = threadIdx.x; j < len; j += 256) sum += __expf(sr[j] - mx);
  red[threadIdx.x] = sum; __syncthreads();
  for (int o = 128; o; o >>= 1) { if (threadIdx.x < o) red[threadIdx.x] += red[threadIdx.x + o]; __syncthreads(); }
  const float inv = 1.0f / red[0];
  for (int j = threadIdx.x; j < pad; j += 256)
    pr[j] = (j < len) ? (bf16_t)(__expf(sr[j] - mx) * inv) : (bf16_t)0.0f;
}

// ------------------------------- misc --------------------------------------
__global__ void wconv_kernel(const float* __restrict__ src, bf16_t* __restrict__ dst,
                             int K, int N)   // src KxN -> dst NxK (bf16)
{
  const long i = (long)blockIdx.x * 256 + threadIdx.x;
  if (i >= (long)K * N) return;
  const long n = i / K, k = i - n * K;
  dst[i] = (bf16_t)src[k * (long)N + n];
}

__global__ void cvt_kernel(const float* __restrict__ s, bf16_t* __restrict__ d, long n)
{
  const long i = (long)blockIdx.x * 256 + threadIdx.x;
  if (i < n) d[i] = (bf16_t)s[i];
}

// V (B,HW,C) -> V^T (B,NH,DH,HW)
__global__ void vtrans_kernel(const bf16_t* __restrict__ v, bf16_t* __restrict__ vt)
{
  const long i = (long)blockIdx.x * 256 + threadIdx.x;
  if (i >= (long)BB * CC * HWD) return;
  const long j = i & (HWD - 1);
  const long d = (i >> 10) & (DHD - 1);
  const long h = (i >> 16) & (NHD - 1);
  const long b = i >> 19;
  vt[i] = v[(b * HWD + j) * CC + h * DHD + d];
}

// Vc (B,NCP,C) -> Vc^T (B,NH,DH,NCS), zero-padding keys [NCD, NCS)
__global__ void vtransc_kernel(const bf16_t* __restrict__ vc, bf16_t* __restrict__ vtc)
{
  const long i = (long)blockIdx.x * 256 + threadIdx.x;
  if (i >= (long)BB * NHD * DHD * NCS) return;
  const int  kk = (int)(i % NCS);
  const long r  = i / NCS;
  const long d  = r & (DHD - 1);
  const long h  = (r >> 6) & (NHD - 1);
  const long b  = r >> 9;
  vtc[i] = (kk < NCD) ? vc[(b * NCP + kk) * CC + h * DHD + d] : (bf16_t)0.0f;
}

__global__ void geglu_kernel(const bf16_t* __restrict__ hg, bf16_t* __restrict__ gg, long rows)
{
  const long i = (long)blockIdx.x * 256 + threadIdx.x;
  if (i >= rows * F2D) return;
  const long r = i >> 11;
  const int  c = (int)(i & (F2D - 1));
  const float u = (float)hg[r * F1D + c];
  const float g = (float)hg[r * F1D + F2D + c];
  const float ge = 0.5f * g * (1.0f + erff(g * 0.70710678118654752f));
  gg[i] = (bf16_t)(u * ge);
}

__global__ void finaladd_kernel(const float* __restrict__ yt, const float* __restrict__ x0,
                                float* __restrict__ out)
{
  const long i = (long)blockIdx.x * 256 + threadIdx.x;
  if (i >= (long)BB * CC * HWD) return;
  const long p = i & (HWD - 1);
  const long c = (i >> 10) & (CC - 1);
  const long b = i >> 19;
  out[i] = yt[((b * HWD + p) * CC) + c] + x0[i];
}

// ------------------------------- host --------------------------------------
template<bool F32, bool ACC>
static inline void gemm(hipStream_t st,
                        const bf16_t* A, long lda, long sA,
                        const bf16_t* BT, long ldb, long sB,
                        void* D, long ldd, long sD, const float* bias,
                        int M, int N, int Kv, int nz)
{
  const int tiles = ((M + 31) / 32) * ((N + 31) / 32);
  dim3 grid((tiles + 7) / 8, 1, nz);
  gemm_wmma<F32, ACC><<<grid, 256, 0, st>>>(A, lda, sA, BT, ldb, sB, D, ldd, sD, bias, M, N, Kv);
}

extern "C" void kernel_launch(void* const* d_in, const int* in_sizes, int n_in,
                              void* d_out, int out_size, void* d_ws, size_t ws_size,
                              hipStream_t stream)
{
  (void)in_sizes; (void)n_in; (void)out_size; (void)ws_size;
  const float* x     = (const float*)d_in[0];
  const float* cond  = (const float*)d_in[1];
  const float* gn_s  = (const float*)d_in[2];
  const float* gn_b  = (const float*)d_in[3];
  const float* pin_w = (const float*)d_in[4];
  const float* pin_b = (const float*)d_in[5];
  const float* pout_w= (const float*)d_in[6];
  const float* pout_b= (const float*)d_in[7];
  const float* ln1_s = (const float*)d_in[8];
  const float* ln1_b = (const float*)d_in[9];
  const float* sa_wq = (const float*)d_in[10];
  const float* sa_wk = (const float*)d_in[11];
  const float* sa_wv = (const float*)d_in[12];
  const float* sa_wo = (const float*)d_in[13];
  const float* sa_bo = (const float*)d_in[14];
  const float* ln2_s = (const float*)d_in[15];
  const float* ln2_b = (const float*)d_in[16];
  const float* ca_wq = (const float*)d_in[17];
  const float* ca_wk = (const float*)d_in[18];
  const float* ca_wv = (const float*)d_in[19];
  const float* ca_wo = (const float*)d_in[20];
  const float* ca_bo = (const float*)d_in[21];
  const float* ln3_s = (const float*)d_in[22];
  const float* ln3_b = (const float*)d_in[23];
  const float* ff_w1 = (const float*)d_in[24];
  const float* ff_b1 = (const float*)d_in[25];
  const float* ff_w2 = (const float*)d_in[26];
  const float* ff_b2 = (const float*)d_in[27];

  // ---- workspace bump allocator (~170 MB total) ----
  char* cur = (char*)d_ws;
  auto bump = [&](size_t bytes) -> void* {
    void* r = (void*)cur; cur += (bytes + 255) & ~(size_t)255; return r;
  };
  bf16_t* wT_pin  = (bf16_t*)bump((size_t)CC * CC * 2);
  bf16_t* wT_pout = (bf16_t*)bump((size_t)CC * CC * 2);
  bf16_t *wsaq[LL], *wsak[LL], *wsav[LL], *wsao[LL];
  bf16_t *wcaq[LL], *wcak[LL], *wcav[LL], *wcao[LL], *wf1[LL], *wf2[LL];
  for (int i = 0; i < LL; ++i) {
    wsaq[i] = (bf16_t*)bump((size_t)CC * CC * 2);
    wsak[i] = (bf16_t*)bump((size_t)CC * CC * 2);
    wsav[i] = (bf16_t*)bump((size_t)CC * CC * 2);
    wsao[i] = (bf16_t*)bump((size_t)CC * CC * 2);
    wcaq[i] = (bf16_t*)bump((size_t)CC * CC * 2);
    wcak[i] = (bf16_t*)bump((size_t)CC * DCD * 2);
    wcav[i] = (bf16_t*)bump((size_t)CC * DCD * 2);
    wcao[i] = (bf16_t*)bump((size_t)CC * CC * 2);
    wf1[i]  = (bf16_t*)bump((size_t)F1D * CC * 2);
    wf2[i]  = (bf16_t*)bump((size_t)CC * F2D * 2);
  }
  bf16_t* condb  = (bf16_t*)bump((size_t)BB * NCD * DCD * 2);
  float*  t      = (float*) bump((size_t)MALL * CC * 4);
  bf16_t* tn     = (bf16_t*)bump((size_t)MALL * CC * 2);
  bf16_t* q      = (bf16_t*)bump((size_t)MALL * CC * 2);
  bf16_t* k      = (bf16_t*)bump((size_t)MALL * CC * 2);
  bf16_t* v      = (bf16_t*)bump((size_t)MALL * CC * 2);
  bf16_t* vt     = (bf16_t*)bump((size_t)MALL * CC * 2);          // V^T (B,NH,DH,HW)
  bf16_t* kc     = (bf16_t*)bump((size_t)BB * NCP * CC * 2);
  bf16_t* vc     = (bf16_t*)bump((size_t)BB * NCP * CC * 2);
  bf16_t* vtc    = (bf16_t*)bump((size_t)BB * NHD * DHD * NCS * 2); // Vc^T zero-padded
  float*  scores = (float*) bump((size_t)NHD * HWD * HWD * 4);    // per-batch reuse
  bf16_t* probs  = (bf16_t*)bump((size_t)NHD * HWD * HWD * 2);
  bf16_t* ao     = (bf16_t*)bump((size_t)MALL * CC * 2);
  bf16_t* hg     = (bf16_t*)bump((size_t)HWD * F1D * 2);          // per-batch FF chunk
  bf16_t* gg     = (bf16_t*)bump((size_t)HWD * F2D * 2);
  float*  ytmp   = (float*) bump((size_t)MALL * CC * 4);

  // ---- weight convert + transpose to bf16 (NxK) ----
  auto wc = [&](const float* s, bf16_t* d, int K, int N) {
    const long n = (long)K * N;
    wconv_kernel<<<(unsigned)((n + 255) / 256), 256, 0, stream>>>(s, d, K, N);
  };
  wc(pin_w,  wT_pin,  CC, CC);
  wc(pout_w, wT_pout, CC, CC);
  for (int i = 0; i < LL; ++i) {
    wc(sa_wq + (long)i * CC * CC,  wsaq[i], CC, CC);
    wc(sa_wk + (long)i * CC * CC,  wsak[i], CC, CC);
    wc(sa_wv + (long)i * CC * CC,  wsav[i], CC, CC);
    wc(sa_wo + (long)i * CC * CC,  wsao[i], CC, CC);
    wc(ca_wq + (long)i * CC * CC,  wcaq[i], CC, CC);
    wc(ca_wk + (long)i * DCD * CC, wcak[i], DCD, CC);
    wc(ca_wv + (long)i * DCD * CC, wcav[i], DCD, CC);
    wc(ca_wo + (long)i * CC * CC,  wcao[i], CC, CC);
    wc(ff_w1 + (long)i * CC * F1D, wf1[i],  CC, F1D);
    wc(ff_w2 + (long)i * F2D * CC, wf2[i],  F2D, CC);
  }
  {
    const long n = (long)BB * NCD * DCD;
    cvt_kernel<<<(unsigned)((n + 255) / 256), 256, 0, stream>>>(cond, condb, n);
  }

  // ---- GroupNorm -> tokens (bf16), proj_in -> f32 residual stream t ----
  groupnorm_kernel<<<BB * 32, 256, 0, stream>>>(x, gn_s, gn_b, tn);
  gemm<true, false>(stream, tn, CC, 0, wT_pin, CC, 0, t, CC, 0, pin_b, MALL, CC, CC, 1);

  const long NTOT = (long)MALL * CC;
  for (int i = 0; i < LL; ++i) {
    // ---------------- self attention ----------------
    layernorm_kernel<<<MALL, 256, 0, stream>>>(t, ln1_s + i * CC, ln1_b + i * CC, tn, CC);
    gemm<false, false>(stream, tn, CC, 0, wsaq[i], CC, 0, q, CC, 0, nullptr, MALL, CC, CC, 1);
    gemm<false, false>(stream, tn, CC, 0, wsak[i], CC, 0, k, CC, 0, nullptr, MALL, CC, CC, 1);
    gemm<false, false>(stream, tn, CC, 0, wsav[i], CC, 0, v, CC, 0, nullptr, MALL, CC, CC, 1);
    vtrans_kernel<<<(unsigned)((NTOT + 255) / 256), 256, 0, stream>>>(v, vt);
    for (int b = 0; b < BB; ++b) {
      const long ob = (long)b * HWD * CC;
      gemm<true, false>(stream, q + ob, CC, DHD, k + ob, CC, DHD,
                        scores, HWD, (long)HWD * HWD, nullptr, HWD, HWD, DHD, NHD);
      softmax_kernel<<<NHD * HWD, 256, 0, stream>>>(scores, probs, HWD, HWD, HWD);
      gemm<false, false>(stream, probs, HWD, (long)HWD * HWD,
                         vt + ob, HWD, (long)DHD * HWD,
                         ao + ob, CC, DHD, nullptr, HWD, DHD, HWD, NHD);
    }
    gemm<true, true>(stream, ao, CC, 0, wsao[i], CC, 0, t, CC, 0,
                     sa_bo + i * CC, MALL, CC, CC, 1);   // t += attn_out

    // ---------------- cross attention ----------------
    layernorm_kernel<<<MALL, 256, 0, stream>>>(t, ln2_s + i * CC, ln2_b + i * CC, tn, CC);
    gemm<false, false>(stream, tn, CC, 0, wcaq[i], CC, 0, q, CC, 0, nullptr, MALL, CC, CC, 1);
    gemm<false, false>(stream, condb, DCD, (long)NCD * DCD, wcak[i], DCD, 0,
                       kc, CC, (long)NCP * CC, nullptr, NCD, CC, DCD, BB);
    gemm<false, false>(stream, condb, DCD, (long)NCD * DCD, wcav[i], DCD, 0,
                       vc, CC, (long)NCP * CC, nullptr, NCD, CC, DCD, BB);
    {
      const long n = (long)BB * NHD * DHD * NCS;
      vtransc_kernel<<<(unsigned)((n + 255) / 256), 256, 0, stream>>>(vc, vtc);
    }
    for (int b = 0; b < BB; ++b) {
      const long ob = (long)b * HWD * CC;
      gemm<true, false>(stream, q + ob, CC, DHD, kc + (long)b * NCP * CC, CC, DHD,
                        scores, NCS, (long)HWD * NCS, nullptr, HWD, NCD, DHD, NHD);
      softmax_kernel<<<NHD * HWD, 256, 0, stream>>>(scores, probs, NCD, NCS, NCS);
      gemm<false, false>(stream, probs, NCS, (long)HWD * NCS,
                         vtc + (long)b * NHD * DHD * NCS, NCS, (long)DHD * NCS,
                         ao + ob, CC, DHD, nullptr, HWD, DHD, NCS, NHD);
    }
    gemm<true, true>(stream, ao, CC, 0, wcao[i], CC, 0, t, CC, 0,
                     ca_bo + i * CC, MALL, CC, CC, 1);   // t += cross_out

    // ---------------- GEGLU feed-forward ----------------
    layernorm_kernel<<<MALL, 256, 0, stream>>>(t, ln3_s + i * CC, ln3_b + i * CC, tn, CC);
    for (int b = 0; b < BB; ++b) {
      const long ob = (long)b * HWD * CC;
      gemm<false, false>(stream, tn + ob, CC, 0, wf1[i], CC, 0, hg, F1D, 0,
                         ff_b1 + (long)i * F1D, HWD, F1D, CC, 1);
      geglu_kernel<<<(unsigned)(((long)HWD * F2D + 255) / 256), 256, 0, stream>>>(hg, gg, HWD);
      gemm<true, true>(stream, gg, F2D, 0, wf2[i], F2D, 0, t + ob, CC, 0,
                       ff_b2 + i * CC, HWD, CC, F2D, 1);   // t += ff_out
    }
  }

  // ---- proj_out + outer residual ----
  cvt_kernel<<<(unsigned)((NTOT + 255) / 256), 256, 0, stream>>>(t, tn, NTOT);
  gemm<true, false>(stream, tn, CC, 0, wT_pout, CC, 0, ytmp, CC, 0, pout_b,
                    MALL, CC, CC, 1);
  finaladd_kernel<<<(unsigned)(((long)BB * CC * HWD + 255) / 256), 256, 0, stream>>>(
      ytmp, x, (float*)d_out);
}
